// MultiHeadAttention_24309514895893
// MI455X (gfx1250) — matrix operand order repaired and checked by reading, not compiled, not run on hardware
//
#include <hip/hip_runtime.h>
#include <hip/hip_bf16.h>
#include <cmath>

// ---------------- problem constants (match reference) ----------------
static constexpr int B_       = 2;
static constexpr int S_       = 1024;
static constexpr int DM       = 2048;
static constexpr int HQ_      = 32;
static constexpr int HKV_     = 8;
static constexpr int HD_      = 64;       // DM / HQ
static constexpr int G_       = HQ_ / HKV_;
static constexpr float ROPE_BASE = 10000.0f;

typedef __attribute__((ext_vector_type(16))) __bf16        bf16x16;
typedef __attribute__((ext_vector_type(8)))  float         v8f;
typedef __attribute__((ext_vector_type(4)))  unsigned int  u32x4;
typedef __attribute__((ext_vector_type(8)))  int           i32x8;
typedef __attribute__((ext_vector_type(4)))  int           i32x4;

#if defined(__has_builtin)
#  if __has_builtin(__builtin_amdgcn_tensor_load_to_lds)
#    define HAVE_TDM 1
#  endif
#endif

// ---------------- WMMA fragment loaders (fp32 -> bf16 in regs) ----------------
// A tile 16x32 (M x K), base points at element [0,0], row stride in floats.
// Layout: lanes 0-15 hold M=lane, K=0..15 (contiguous);
//         lanes 16-31 hold M=lane-16, K=16..31.
__device__ __forceinline__ bf16x16 load_a_frag(const float* __restrict__ base,
                                               int row_stride, int lane) {
  int r    = lane & 15;
  int koff = (lane & 16) ? 16 : 0;
  const float* p0 = base + (size_t)r * row_stride + koff;
  const float* p1 = p0 + 8;
  bf16x16 a;
#pragma unroll
  for (int i = 0; i < 8; ++i) a[i]     = (__bf16)p0[i];
#pragma unroll
  for (int i = 0; i < 8; ++i) a[8 + i] = (__bf16)p1[i];
  return a;
}

// B tile 32x16 (K x N) where B_wmma[k,n] = Src[n, k] (Src row-major, ld in floats).
// Lane L<16 holds column N=L, K=0..15 (contiguous); lane L+16 holds N=L, K=16..31.
__device__ __forceinline__ bf16x16 load_bT_frag(const float* __restrict__ base,
                                                int row_stride, int lane) {
  int n    = lane & 15;
  int koff = (lane & 16) ? 16 : 0;
  const float* p = base + (size_t)n * row_stride + koff;
  bf16x16 b;
#pragma unroll
  for (int i = 0; i < 16; ++i) b[i] = (__bf16)p[i];
  return b;
}

// B tile 32x16 natural orientation: B_wmma[k,n] = Src[k, n] (row-major rows = K).
__device__ __forceinline__ bf16x16 load_b_frag(const float* __restrict__ base,
                                               int row_stride, int lane) {
  int n    = lane & 15;
  int koff = (lane & 16) ? 16 : 0;
  bf16x16 b;
#pragma unroll
  for (int i = 0; i < 16; ++i)
    b[i] = (__bf16)base[(size_t)(koff + i) * row_stride + n];
  return b;
}

__device__ __forceinline__ v8f wmma_bf16(bf16x16 a, bf16x16 b, v8f c) {
  return __builtin_amdgcn_wmma_f32_16x16x32_bf16(false, a, false, b,
                                                 (short)0, c, false, false);
}

#if defined(HAVE_TDM)
// ---------------- TDM 2D tile load: global (row-major) -> LDS (packed) -----
// D# packing per CDNA5 ISA.
// Tile: rows x 32 fp32 elements, tensor row stride = ld (elements).
// LDS result: rows packed contiguously, 32 floats (128B) per row.
__device__ __forceinline__ void tdm_load_2d(const float* gptr, unsigned lds_addr,
                                            int rows, int ld, int tot_rows) {
  unsigned long long ga = (unsigned long long)(uintptr_t)gptr;
  u32x4 g0;
  g0[0] = 1u;                                              // count=1 (valid)
  g0[1] = lds_addr;                                        // lds_addr bytes
  g0[2] = (unsigned)ga;                                    // global_addr[31:0]
  g0[3] = (unsigned)((ga >> 32) & 0x1ffffffu) | (2u << 30);// addr[56:32] | type=2
  i32x8 g1;
  g1[0] = (int)(2u << 16);                                 // data_size=4B
  g1[1] = (int)(((unsigned)ld & 0xffffu) << 16);           // tensor_dim0[15:0]
  g1[2] = (int)((((unsigned)ld >> 16) & 0xffffu) |
                (((unsigned)tot_rows & 0xffffu) << 16));   // dim0 hi | dim1 lo
  g1[3] = (int)((((unsigned)tot_rows >> 16) & 0xffffu) |
                (32u << 16));                              // dim1 hi | tile_dim0=32
  g1[4] = rows;                                            // tile_dim1 (tile_dim2=0)
  g1[5] = ld;                                              // tensor_dim0_stride[31:0]
  g1[6] = 0;                                               // stride hi | dim1_stride lo
  g1[7] = 0;
  i32x4 gz = {};
#if __clang_major__ >= 23
  i32x8 gz8 = {};
  __builtin_amdgcn_tensor_load_to_lds(g0, g1, gz, gz, gz8, 0);
#else
  __builtin_amdgcn_tensor_load_to_lds(g0, g1, gz, gz, 0);
#endif
}
#endif

// ---------------- GEMM: C[M,N] = A[M,K] * W[N,K]^T + bias[N] ----------------
// Block: 256 threads = 8 waves, waves tiled 4(M) x 2(N), each wave 32x32.
// Block tile 128(M) x 64(N); K-step 32 staged in LDS via TDM, double-buffered.
__global__ __launch_bounds__(256) void gemm_bias_kernel(
    const float* __restrict__ A, const float* __restrict__ W,
    const float* __restrict__ bias, float* __restrict__ C,
    int M, int N, int K) {
  __shared__ float Abuf[2][128][32];
  __shared__ float Bbuf[2][64][32];

  int lane = threadIdx.x & 31;
  int wave = threadIdx.x >> 5;
  int wm   = wave & 3;
  int wn   = wave >> 2;
  int mblk = blockIdx.y * 128;
  int nblk = blockIdx.x * 64;

#if defined(HAVE_TDM)
  unsigned ldsA = (unsigned)(uintptr_t)&Abuf[0][0][0];
  unsigned ldsB = (unsigned)(uintptr_t)&Bbuf[0][0][0];
  if (wave == 0) {  // prologue DMA into buffer 0
    tdm_load_2d(A + (size_t)mblk * K, ldsA, 128, K, M);
    tdm_load_2d(W + (size_t)nblk * K, ldsB, 64,  K, N);
  }
#endif

  v8f acc[2][2] = {};
  int buf = 0;
  for (int k0 = 0; k0 < K; k0 += 32) {
#if defined(HAVE_TDM)
    if (wave == 0) __builtin_amdgcn_s_wait_tensorcnt(0);
    __syncthreads();
    if (wave == 0 && k0 + 32 < K) {  // prefetch next K-step into other buffer
      int nb = buf ^ 1;
      tdm_load_2d(A + (size_t)mblk * K + (k0 + 32), ldsA + (unsigned)nb * 128 * 32 * 4,
                  128, K, M);
      tdm_load_2d(W + (size_t)nblk * K + (k0 + 32), ldsB + (unsigned)nb * 64 * 32 * 4,
                  64, K, N);
    }
#else
    // Fallback: cooperative copy global -> LDS (no TDM builtin on this toolchain)
    for (int i = threadIdx.x; i < 128 * 32; i += 256)
      Abuf[buf][i >> 5][i & 31] = A[(size_t)(mblk + (i >> 5)) * K + k0 + (i & 31)];
    for (int i = threadIdx.x; i < 64 * 32; i += 256)
      Bbuf[buf][i >> 5][i & 31] = W[(size_t)(nblk + (i >> 5)) * K + k0 + (i & 31)];
    __syncthreads();
#endif
    const float(*As)[32] = Abuf[buf];
    const float(*Bs)[32] = Bbuf[buf];
    bf16x16 a0 = load_a_frag (&As[wm * 32 +  0][0], 32, lane);
    bf16x16 a1 = load_a_frag (&As[wm * 32 + 16][0], 32, lane);
    bf16x16 b0 = load_bT_frag(&Bs[wn * 32 +  0][0], 32, lane);
    bf16x16 b1 = load_bT_frag(&Bs[wn * 32 + 16][0], 32, lane);
    acc[0][0] = wmma_bf16(a0, b0, acc[0][0]);
    acc[0][1] = wmma_bf16(a0, b1, acc[0][1]);
    acc[1][0] = wmma_bf16(a1, b0, acc[1][0]);
    acc[1][1] = wmma_bf16(a1, b1, acc[1][1]);
    __syncthreads();
    buf ^= 1;
  }

  int m0 = mblk + wm * 32;
  int n0 = nblk + wn * 32;
  int mrow = (lane & 16) ? 8 : 0;
#pragma unroll
  for (int mi = 0; mi < 2; ++mi) {
#pragma unroll
    for (int ni = 0; ni < 2; ++ni) {
      int n = n0 + ni * 16 + (lane & 15);
      float bv = bias[n];
#pragma unroll
      for (int i = 0; i < 8; ++i) {
        int m = m0 + mi * 16 + mrow + i;
        C[(size_t)m * N + n] = acc[mi][ni][i] + bv;
      }
    }
  }
}

// ---------------- RoPE (in place). x layout (B, S, H, HD). ----------------
__global__ __launch_bounds__(256) void rope_kernel(
    float* __restrict__ x, const int* __restrict__ pos_ids,
    int H, int total_rows /* B*S*H */) {
  constexpr int HALF = HD_ / 2;      // 32
  int idx = blockIdx.x * blockDim.x + threadIdx.x;
  if (idx >= total_rows * HALF) return;
  int d   = idx & (HALF - 1);
  int row = idx >> 5;                // /HALF
  int bs  = row / H;                 // b*S + s
  float pos = (float)pos_ids[bs];
  float inv_freq = __expf(-(2.0f * (float)d / (float)HD_) * __logf(ROPE_BASE));
  float ang = pos * inv_freq;
  float c = __cosf(ang), sn = __sinf(ang);
  float* p = x + (size_t)row * HD_;
  float x1 = p[d], x2 = p[d + HALF];
  p[d]        = x1 * c - x2 * sn;
  p[d + HALF] = x2 * c + x1 * sn;
}

// ---------------- scores = Q K^T / sqrt(HD), causal masked -----------------
// grid: (S/16, B*HQ). 8 waves stripe over causal k-tiles of one 16-row q tile.
// qbuf layout (B,S,HQ,HD); kbuf layout (B,S,HKV,HD).
__global__ __launch_bounds__(256) void scores_kernel(
    const float* __restrict__ qbuf, const float* __restrict__ kbuf,
    float* __restrict__ attn_w) {
  int lane = threadIdx.x & 31;
  int wave = threadIdx.x >> 5;
  int qt = blockIdx.x;
  int bh = blockIdx.y;
  int b  = bh / HQ_;
  int h  = bh % HQ_;
  int hkv = h / G_;
  int q0 = qt * 16;

  const float* qbase = qbuf + (((size_t)b * S_ + q0) * HQ_ + h) * HD_;
  bf16x16 a0 = load_a_frag(qbase +  0, HQ_ * HD_, lane);
  bf16x16 a1 = load_a_frag(qbase + 32, HQ_ * HD_, lane);

  float* orow = attn_w + ((size_t)bh * S_ + q0) * S_;
  int mrow = (lane & 16) ? 8 : 0;
  int ktiles = qt + 1;               // causal: only tiles with k_base <= q0+15
  for (int kt = wave; kt < ktiles; kt += 8) {
    int k0 = kt * 16;
    const float* kbase = kbuf + (((size_t)b * S_ + k0) * HKV_ + hkv) * HD_;
    bf16x16 b0 = load_bT_frag(kbase +  0, HKV_ * HD_, lane);
    bf16x16 b1 = load_bT_frag(kbase + 32, HKV_ * HD_, lane);
    v8f acc = {};
    acc = wmma_bf16(a0, b0, acc);
    acc = wmma_bf16(a1, b1, acc);
    int kp = k0 + (lane & 15);
#pragma unroll
    for (int i = 0; i < 8; ++i) {
      int qp = q0 + mrow + i;
      float v = acc[i] * 0.125f;     // 1/sqrt(64)
      if (kp > qp) v = -3.4028235e38f;
      orow[(size_t)(mrow + i) * S_ + kp] = v;
    }
  }
}

// ---------------- causal row softmax over attn_w, zero masked tail ---------
__global__ __launch_bounds__(256) void softmax_kernel(float* __restrict__ attn_w) {
  int r = blockIdx.x;                // row index in (B*HQ*S)
  int q = r % S_;
  int valid = q + 1;
  float* row = attn_w + (size_t)r * S_;
  int t = threadIdx.x;
  __shared__ float red[256];

  float m = -INFINITY;
  for (int j = t; j < valid; j += 256) m = fmaxf(m, row[j]);
  red[t] = m; __syncthreads();
  for (int s = 128; s > 0; s >>= 1) {
    if (t < s) red[t] = fmaxf(red[t], red[t + s]);
    __syncthreads();
  }
  m = red[0]; __syncthreads();

  float sum = 0.0f;
  for (int j = t; j < valid; j += 256) {
    float e = __expf(row[j] - m);
    row[j] = e;
    sum += e;
  }
  red[t] = sum; __syncthreads();
  for (int s = 128; s > 0; s >>= 1) {
    if (t < s) red[t] += red[t + s];
    __syncthreads();
  }
  float inv = 1.0f / red[0];
  for (int j = t; j < valid; j += 256) row[j] *= inv;
  for (int j = valid + t; j < S_; j += 256) row[j] = 0.0f;  // masked tail = 0
}

// ---------------- ctx = P * V  (per head, 16 q-rows x 64 cols) -------------
// grid: (S/16, B*HQ), 1 wave per block. ctx layout (B,S,HQ,HD).
__global__ __launch_bounds__(32) void ctx_kernel(
    const float* __restrict__ attn_w, const float* __restrict__ vbuf,
    float* __restrict__ ctx) {
  int lane = threadIdx.x & 31;
  int qt = blockIdx.x;
  int bh = blockIdx.y;
  int b  = bh / HQ_;
  int h  = bh % HQ_;
  int hkv = h / G_;
  int q0 = qt * 16;

  const float* arow = attn_w + ((size_t)bh * S_ + q0) * S_;  // 16 rows, ld = S
  v8f acc[4] = {};
  int kmax = q0 + 16;                // weights beyond the diagonal are exact 0
  for (int k0 = 0; k0 < kmax; k0 += 32) {
    bf16x16 a = load_a_frag(arow + k0, S_, lane);
    const float* vb = vbuf + (((size_t)b * S_ + k0) * HKV_ + hkv) * HD_;
#pragma unroll
    for (int nt = 0; nt < 4; ++nt) {
      bf16x16 bv = load_b_frag(vb + nt * 16, HKV_ * HD_, lane);
      acc[nt] = wmma_bf16(a, bv, acc[nt]);
    }
  }
  int mrow = (lane & 16) ? 8 : 0;
#pragma unroll
  for (int nt = 0; nt < 4; ++nt) {
    int hd = nt * 16 + (lane & 15);
#pragma unroll
    for (int i = 0; i < 8; ++i) {
      int qp = q0 + mrow + i;
      ctx[(((size_t)b * S_ + qp) * HQ_ + h) * HD_ + hd] = acc[nt][i];
    }
  }
}

// ---------------- launcher ----------------
extern "C" void kernel_launch(void* const* d_in, const int* in_sizes, int n_in,
                              void* d_out, int out_size, void* d_ws, size_t ws_size,
                              hipStream_t stream) {
  const float* query  = (const float*)d_in[0];
  const float* key    = (const float*)d_in[1];
  const float* value  = (const float*)d_in[2];
  // d_in[3] = attn_mask (bool causal) — implied by construction, unused.
  const int*   pos    = (const int*)d_in[4];
  const float* q_w    = (const float*)d_in[5];
  const float* q_b    = (const float*)d_in[6];
  const float* k_w    = (const float*)d_in[7];
  const float* k_b    = (const float*)d_in[8];
  const float* v_w    = (const float*)d_in[9];
  const float* v_b    = (const float*)d_in[10];
  const float* o_w    = (const float*)d_in[11];
  const float* o_b    = (const float*)d_in[12];

  float* attn_out = (float*)d_out;                                   // (B,S,DM)
  float* attn_w   = (float*)d_out + (size_t)B_ * S_ * DM;            // (B,HQ,S,S)

  // workspace: qbuf | kbuf | vbuf | ctx  (all fp32)
  float* qbuf = (float*)d_ws;                                        // B*S*HQ*HD
  float* kbuf = qbuf + (size_t)B_ * S_ * HQ_  * HD_;                 // B*S*HKV*HD
  float* vbuf = kbuf + (size_t)B_ * S_ * HKV_ * HD_;
  float* ctx  = vbuf + (size_t)B_ * S_ * HKV_ * HD_;

  const int M = B_ * S_;             // 2048

  // 1) Q/K/V projections (output naturally lands in (B,S,H,HD) layout)
  {
    dim3 blk(256);
    dim3 gq((HQ_ * HD_) / 64, M / 128);
    gemm_bias_kernel<<<gq, blk, 0, stream>>>(query, q_w, q_b, qbuf, M, HQ_ * HD_, DM);
    dim3 gk((HKV_ * HD_) / 64, M / 128);
    gemm_bias_kernel<<<gk, blk, 0, stream>>>(key,   k_w, k_b, kbuf, M, HKV_ * HD_, DM);
    gemm_bias_kernel<<<gk, blk, 0, stream>>>(value, v_w, v_b, vbuf, M, HKV_ * HD_, DM);
  }

  // 2) RoPE on q and k
  {
    int qrows = B_ * S_ * HQ_;
    int krows = B_ * S_ * HKV_;
    rope_kernel<<<(qrows * (HD_ / 2) + 255) / 256, 256, 0, stream>>>(qbuf, pos, HQ_,  qrows);
    rope_kernel<<<(krows * (HD_ / 2) + 255) / 256, 256, 0, stream>>>(kbuf, pos, HKV_, krows);
  }

  // 3) masked scores -> attn_w region of d_out
  scores_kernel<<<dim3(S_ / 16, B_ * HQ_), 256, 0, stream>>>(qbuf, kbuf, attn_w);

  // 4) causal softmax in place (also zeroes masked tail)
  softmax_kernel<<<B_ * HQ_ * S_, 256, 0, stream>>>(attn_w);

  // 5) ctx = P * V, stored as (B,S,HQ,HD) == (B,S,DM)
  ctx_kernel<<<dim3(S_ / 16, B_ * HQ_), 32, 0, stream>>>(attn_w, vbuf, ctx);

  // 6) output projection
  gemm_bias_kernel<<<dim3(DM / 64, M / 128), 256, 0, stream>>>(ctx, o_w, o_b, attn_out, M, DM, DM);
}
